// BipartiteNeuralMessagePassingLayer_83210696393444
// MI455X (gfx1250) — matrix operand order repaired and checked by reading, not compiled, not run on hardware
//
#include <hip/hip_runtime.h>

#if __has_include(<hip/amd_detail/amd_gfx1250_TDM.h>)
#define TDM_ARITY6 1   // amdgpu-toolchain (clang-23): 6-arg tensor builtin
#endif

// ---------- types ----------
typedef __attribute__((ext_vector_type(16))) __bf16         v16bf;
typedef __attribute__((ext_vector_type(8)))  __bf16         v8bf;
typedef __attribute__((ext_vector_type(8)))  float          v8f;
typedef __attribute__((ext_vector_type(8)))  unsigned short v8us;
typedef __attribute__((ext_vector_type(4)))  unsigned int   u32x4;
typedef __attribute__((ext_vector_type(8)))  int            i32x8;
typedef __attribute__((ext_vector_type(4)))  int            i32x4;

union Frag {
    struct { v8bf lo, hi; } b;
    struct { v8us lo, hi; } u;
    v16bf v;
};

__device__ __forceinline__ unsigned short f2bf(float f) {
    unsigned int u = __float_as_uint(f);
    unsigned int r = u + 0x7FFFu + ((u >> 16) & 1u);   // round-to-nearest-even
    return (unsigned short)(r >> 16);
}

#define ND 128
#define NA 512
#define NB 512
#define EL_ELEMS (33554432LL)   // 512*512*128

// ---------------------------------------------------------------------------
// prep: transpose-convert weights to bf16, N-major:
//   WeT [n][k] (128 x 256)  = We1[256+k][n]
//   We2T[n][k] (128 x 128)  = We2[k][n]
// ---------------------------------------------------------------------------
__global__ __launch_bounds__(256) void prep_kernel(
    const float* __restrict__ We1, const float* __restrict__ We2,
    unsigned short* __restrict__ WeT, unsigned short* __restrict__ We2T)
{
    int i = blockIdx.x * 256 + threadIdx.x;
    if (i < 32768) {
        int n = i >> 8, k = i & 255;
        WeT[i] = f2bf(We1[(256 + k) * ND + n]);
    } else if (i < 49152) {
        int j = i - 32768;
        int n = j >> 7, k = j & 127;
        We2T[j] = f2bf(We2[k * ND + n]);
    }
}

// ---------------------------------------------------------------------------
// proj: rowA = nodes_a @ We1[:128], rowB = nodes_b @ We1[128:256]   (f32)
// ---------------------------------------------------------------------------
__global__ __launch_bounds__(128) void proj_kernel(
    const float* __restrict__ nodes_a, const float* __restrict__ nodes_b,
    const float* __restrict__ We1,
    float* __restrict__ rowA, float* __restrict__ rowB)
{
    __shared__ float x[ND];
    int r = blockIdx.x;          // 0..1023
    int n = threadIdx.x;
    const float* src = (r < NA) ? (nodes_a + r * ND) : (nodes_b + (r - NA) * ND);
    x[n] = src[n];
    __syncthreads();
    const float* W = (r < NA) ? We1 : (We1 + ND * ND);
    float s = 0.f;
    #pragma unroll 8
    for (int k = 0; k < ND; ++k) s += x[k] * W[k * ND + n];
    if (r < NA) rowA[r * ND + n] = s;
    else        rowB[(r - NA) * ND + n] = s;
}

// ---------------------------------------------------------------------------
// edge: per block = 64 edge-rows (fixed a, 64 consecutive b) x 128 outputs.
// TDM stages the contiguous 64x256 f32 tile into LDS; GEMM1 (K=256, bf16
// WMMA with cvt-on-load) + bias/ReLU -> LDS bf16 h -> GEMM2 (K=128) ->
// bias/ReLU -> edge_latent.
// 8 waves: wave w -> M-subtile (w>>1)*16, N-half (w&1)*64 (4x 16-col tiles).
// ---------------------------------------------------------------------------
__global__ __launch_bounds__(256) void edge_kernel(
    const float* __restrict__ edge_embeds,
    const float* __restrict__ rowA, const float* __restrict__ rowB,
    const float* __restrict__ be1,  const float* __restrict__ be2,
    const unsigned short* __restrict__ WeT,
    const unsigned short* __restrict__ We2T,
    float* __restrict__ edge_latent)
{
    __shared__ __align__(32) float          sEdge[64][256];  // 64 KB (TDM dest)
    __shared__ __align__(16) unsigned short sH[64][128];     // 16 KB

    const int t   = threadIdx.x;
    const int blk = blockIdx.x;                 // 4096
    const int a   = blk >> 3;
    const int b0  = (blk & 7) * 64;
    const long long rowBase = (long long)a * NB + b0;

    // ---- TDM async copy: 64x256 f32 tile (fully contiguous, 64 KB) -> LDS --
    if (t < 32) {                               // wave 0 issues the descriptor
        // LDS byte offset = low 32 bits of the flat address (ISA 10.2).
        unsigned lds_off = (unsigned)(unsigned long long)(const void*)&sEdge[0][0];
        unsigned long long ga =
            (unsigned long long)(const void*)(edge_embeds + rowBase * 256);

        u32x4 g0 = {};
        g0.x = 1u;                              // count=1, user descriptor
        g0.y = lds_off;                         // lds_addr [63:32]
        g0.z = (unsigned)ga;                    // global_addr lo [95:64]
        g0.w = (unsigned)((ga >> 32) & 0x1FFFFFFu) | (2u << 30); // hi + type=2

        i32x8 g1 = {};
        g1[0] = (2 << 16);                      // data_size = 4 bytes
        g1[1] = (int)(16384u << 16);            // tensor_dim0[15:0]=16384 @ [63:48]
        g1[2] = (1 << 16);                      // tensor_dim1 = 1      @ [95:80]
        g1[3] = (int)(16384u << 16);            // tile_dim0 = 16384    @ [127:112]
        g1[4] = 1;                              // tile_dim1 = 1        @ [143:128]
        g1[5] = 16384;                          // tensor_dim0_stride   @ [191:160]
        // g1[6], g1[7] = 0

        i32x4 g2 = {}, g3 = {};                 // 1-D tensor: groups 2/3 unused
#ifdef TDM_ARITY6
        i32x8 g4 = {};
        __builtin_amdgcn_tensor_load_to_lds(g0, g1, g2, g3, g4, 0);
#else
        __builtin_amdgcn_tensor_load_to_lds(g0, g1, g2, g3, 0);
#endif
        __builtin_amdgcn_s_wait_tensorcnt(0);
    }
    __syncthreads();

    const int lane  = t & 31;
    const int lh    = lane >> 4;                // lane half
    const int ln    = lane & 15;
    const int mtile = ((t >> 5) >> 1) * 16;     // wave's M base
    const int ngrp  = ((t >> 5) & 1) * 64;      // wave's N base

    v8f acc[4];
    #pragma unroll
    for (int nt = 0; nt < 4; ++nt) acc[nt] = {};

    // -------- GEMM1: h = edge_tile(64x256) x We(256x128) --------
    for (int k = 0; k < 256; k += 32) {
        Frag af;                                // A: 16x32 bf16 layout
        const int ar = mtile + ln;
        const int ak = k + lh * 8;
        af.b.lo = __builtin_convertvector(*(const v8f*)&sEdge[ar][ak],      v8bf);
        af.b.hi = __builtin_convertvector(*(const v8f*)&sEdge[ar][ak + 16], v8bf);
        #pragma unroll
        for (int nt = 0; nt < 4; ++nt) {
            Frag bf;                            // B: 32x16 bf16 layout (table K order)
            const unsigned short* bp = WeT + (ngrp + nt * 16 + ln) * 256 + (k + lh * 8);
            bf.u.lo = *(const v8us*)bp;
            bf.u.hi = *(const v8us*)(bp + 16);
            acc[nt] = __builtin_amdgcn_wmma_f32_16x16x32_bf16(
                false, af.v, false, bf.v, (short)0, acc[nt], false, false);
        }
    }

    // -------- epilogue 1: + rowA[a] + rowB[b] + be1, ReLU -> sH (bf16) -----
    #pragma unroll
    for (int nt = 0; nt < 4; ++nt) {
        const int n = ngrp + nt * 16 + ln;
        const float bias = rowA[a * ND + n] + be1[n];
        #pragma unroll
        for (int v = 0; v < 8; ++v) {
            const int m = mtile + lh * 8 + v;
            float h = acc[nt][v] + bias + rowB[(b0 + m) * ND + n];
            h = h > 0.f ? h : 0.f;
            sH[m][n] = f2bf(h);
        }
    }
    __syncthreads();

    // -------- GEMM2: el = sH(64x128) x We2(128x128) --------
    #pragma unroll
    for (int nt = 0; nt < 4; ++nt) acc[nt] = {};
    for (int k = 0; k < 128; k += 32) {
        Frag af;
        const int ar = mtile + ln;
        const int ak = k + lh * 8;
        af.u.lo = *(const v8us*)&sH[ar][ak];
        af.u.hi = *(const v8us*)&sH[ar][ak + 16];
        #pragma unroll
        for (int nt = 0; nt < 4; ++nt) {
            Frag bf;
            const unsigned short* bp = We2T + (ngrp + nt * 16 + ln) * 128 + (k + lh * 8);
            bf.u.lo = *(const v8us*)bp;
            bf.u.hi = *(const v8us*)(bp + 16);
            acc[nt] = __builtin_amdgcn_wmma_f32_16x16x32_bf16(
                false, af.v, false, bf.v, (short)0, acc[nt], false, false);
        }
    }

    // -------- epilogue 2: + be2, ReLU -> global edge_latent (f32) ----------
    #pragma unroll
    for (int nt = 0; nt < 4; ++nt) {
        const int n = ngrp + nt * 16 + ln;
        const float bias = be2[n];
        #pragma unroll
        for (int v = 0; v < 8; ++v) {
            const int m = mtile + lh * 8 + v;
            float e = acc[nt][v] + bias;
            e = e > 0.f ? e : 0.f;
            edge_latent[(rowBase + m) * ND + n] = e;
        }
    }
}

// ---------------------------------------------------------------------------
// reduce: sum_a[a] = sum_b EL[a,b,:] ; sum_b[b] = sum_a EL[a,b,:]
// (edge_latent ~134 MB -> resident in 192 MB L2, re-reads are cheap)
// ---------------------------------------------------------------------------
__global__ __launch_bounds__(128) void reduce_kernel(
    const float* __restrict__ EL,
    float* __restrict__ sum_a, float* __restrict__ sum_b)
{
    int blk = blockIdx.x;       // 0..1023
    int n   = threadIdx.x;
    if (blk < NA) {
        const float* p = EL + (long long)blk * NB * ND + n;
        float s = 0.f;
        for (int b = 0; b < NB; ++b) s += p[(long long)b * ND];
        sum_a[blk * ND + n] = s;
    } else {
        int b = blk - NA;
        const float* p = EL + (long long)b * ND + n;
        float s = 0.f;
        for (int a = 0; a < NA; ++a) s += p[(long long)a * NB * ND];
        sum_b[b * ND + n] = s;
    }
}

// ---------------------------------------------------------------------------
// node MLP: relu(relu([emb, sum] @ Wn1 + bn1) @ Wn2 + bn2)
// ---------------------------------------------------------------------------
__global__ __launch_bounds__(128) void node_kernel(
    const float* __restrict__ nodes_a, const float* __restrict__ nodes_b,
    const float* __restrict__ sum_a,   const float* __restrict__ sum_b,
    const float* __restrict__ Wn1, const float* __restrict__ bn1,
    const float* __restrict__ Wn2, const float* __restrict__ bn2,
    float* __restrict__ out_nodes)
{
    __shared__ float x[2 * ND];
    __shared__ float h1[ND];
    int r = blockIdx.x, n = threadIdx.x;
    const float *emb, *sum; float* dst;
    if (r < NA) { emb = nodes_a + r * ND;        sum = sum_a + r * ND;
                  dst = out_nodes + r * ND; }
    else        { int rb = r - NA;
                  emb = nodes_b + rb * ND;       sum = sum_b + rb * ND;
                  dst = out_nodes + NA * ND + rb * ND; }
    x[n] = emb[n]; x[ND + n] = sum[n];
    __syncthreads();
    float s = bn1[n];
    #pragma unroll 8
    for (int k = 0; k < 2 * ND; ++k) s += x[k] * Wn1[k * ND + n];
    h1[n] = s > 0.f ? s : 0.f;
    __syncthreads();
    s = bn2[n];
    #pragma unroll 8
    for (int k = 0; k < ND; ++k) s += h1[k] * Wn2[k * ND + n];
    dst[n] = s > 0.f ? s : 0.f;
}

// ---------------------------------------------------------------------------
extern "C" void kernel_launch(void* const* d_in, const int* in_sizes, int n_in,
                              void* d_out, int out_size, void* d_ws, size_t ws_size,
                              hipStream_t stream) {
    const float* edge_embeds = (const float*)d_in[0];
    const float* nodes_a     = (const float*)d_in[1];
    const float* nodes_b     = (const float*)d_in[2];
    const float* We1         = (const float*)d_in[3];
    const float* be1         = (const float*)d_in[4];
    const float* We2         = (const float*)d_in[5];
    const float* be2         = (const float*)d_in[6];
    const float* Wn1         = (const float*)d_in[7];
    const float* bn1         = (const float*)d_in[8];
    const float* Wn2         = (const float*)d_in[9];
    const float* bn2         = (const float*)d_in[10];

    float* edge_latent = (float*)d_out;                 // 512*512*128
    float* out_nodes   = edge_latent + EL_ELEMS;        // nodes_a then nodes_b

    // workspace layout
    float* rowA  = (float*)d_ws;                        // 512*128
    float* rowB  = rowA + NA * ND;                      // 512*128
    float* sum_a = rowB + NB * ND;                      // 512*128
    float* sum_b = sum_a + NA * ND;                     // 512*128
    unsigned short* WeT  = (unsigned short*)(sum_b + NB * ND);  // 128*256 bf16
    unsigned short* We2T = WeT + 128 * 256;                     // 128*128 bf16

    prep_kernel  <<<192,  256, 0, stream>>>(We1, We2, WeT, We2T);
    proj_kernel  <<<1024, 128, 0, stream>>>(nodes_a, nodes_b, We1, rowA, rowB);
    edge_kernel  <<<4096, 256, 0, stream>>>(edge_embeds, rowA, rowB, be1, be2,
                                            WeT, We2T, edge_latent);
    reduce_kernel<<<1024, 128, 0, stream>>>(edge_latent, sum_a, sum_b);
    node_kernel  <<<1024, 128, 0, stream>>>(nodes_a, nodes_b, sum_a, sum_b,
                                            Wn1, bn1, Wn2, bn2, out_nodes);
}
